// TtAttention_82867099009984
// MI455X (gfx1250) — matrix operand order repaired and checked by reading, not compiled, not run on hardware
//
#include <hip/hip_runtime.h>
#include <hip/hip_bf16.h>

// ---------------------------------------------------------------------------
// MI455X (gfx1250) attention layer: bf16 WMMA GEMMs + flash attention.
// wave32, 16x16x32 bf16 WMMA, f32 accumulation.
// ---------------------------------------------------------------------------

typedef __attribute__((ext_vector_type(16))) __bf16 v16bf;
typedef __attribute__((ext_vector_type(8)))  float  v8f;

#define SEQL   2048
#define DIMD   4096
#define NH     32
#define NKV    8
#define HD     128
#define QDIM   (NH * HD)    // 4096
#define KVDIM  (NKV * HD)   // 1024
#define WIN    4096
#define ATTN_SCALE 0.08838834764831845f   // 128^-0.5

union BFrag { v16bf v; uint4 u[2]; };

__device__ __forceinline__ v8f wmma_bf16(const BFrag& a, const BFrag& b, v8f c) {
    return __builtin_amdgcn_wmma_f32_16x16x32_bf16(
        /*neg_a=*/false, a.v, /*neg_b=*/false, b.v,
        /*c_mod=*/(short)0, c, /*reuse_a=*/false, /*reuse_b=*/false);
}

// ---------------------------------------------------------------------------
// f32 -> bf16 conversion (vectorized x4)
// ---------------------------------------------------------------------------
__global__ void cvt_f32_to_bf16(const float* __restrict__ src,
                                __bf16* __restrict__ dst, int n4) {
    int i = blockIdx.x * blockDim.x + threadIdx.x;
    if (i >= n4) return;
    float4 f = ((const float4*)src)[i];
    __bf16* d = dst + 4 * (size_t)i;
    d[0] = (__bf16)f.x; d[1] = (__bf16)f.y;
    d[2] = (__bf16)f.z; d[3] = (__bf16)f.w;
}

// ---------------------------------------------------------------------------
// C(MxN) = A(MxK) @ B(NxK)^T    A,B bf16 row-major, C f32 or bf16.
// Block: 256 threads (8 waves). Tile 128x128, K-step 32.
// Wave w computes rows [w*16, w*16+16) x all 8 16-col groups.
// ---------------------------------------------------------------------------
template <bool STORE_F32>
__global__ __launch_bounds__(256) void gemm_bf16_xwT(
    const __bf16* __restrict__ A, const __bf16* __restrict__ B,
    void* __restrict__ C, int M, int N, int K)
{
    __shared__ __attribute__((aligned(16))) __bf16 sA[128][32];
    __shared__ __attribute__((aligned(16))) __bf16 sB[128][32];

    const int tid  = threadIdx.x;
    const int wave = tid >> 5;
    const int lane = tid & 31;
    const int lrow = lane & 15;
    const int asel = (lane >> 4) * 8;    // A-matrix K sub-select (16-bit layout)
    const int bsel = (lane >> 4) * 16;   // unused; B loads use asel, same K order as A
    const int m0 = blockIdx.x * 128;
    const int n0 = blockIdx.y * 128;

    v8f acc[8] = {};

    for (int k0 = 0; k0 < K; k0 += 32) {
        __syncthreads();
        // Cooperative stage: 128 rows x 32 cols per matrix = 512 x 16B; 2 per thread.
        #pragma unroll
        for (int p = 0; p < 2; ++p) {
            int idx = tid + p * 256;
            int r = idx >> 2, ch = (idx & 3) * 8;
            *(uint4*)&sA[r][ch] = *(const uint4*)&A[(size_t)(m0 + r) * K + k0 + ch];
            *(uint4*)&sB[r][ch] = *(const uint4*)&B[(size_t)(n0 + r) * K + k0 + ch];
            if (k0 + 32 < K) {   // global_prefetch_b8 of next K tile
                __builtin_prefetch(&A[(size_t)(m0 + r) * K + k0 + 32 + ch], 0, 1);
                __builtin_prefetch(&B[(size_t)(n0 + r) * K + k0 + 32 + ch], 0, 1);
            }
        }
        __syncthreads();

        BFrag af;
        af.u[0] = *(const uint4*)&sA[wave * 16 + lrow][asel];
        af.u[1] = *(const uint4*)&sA[wave * 16 + lrow][16 + asel];
        #pragma unroll
        for (int g = 0; g < 8; ++g) {
            BFrag bf;
            bf.u[0] = *(const uint4*)&sB[g * 16 + lrow][asel];
            bf.u[1] = *(const uint4*)&sB[g * 16 + lrow][16 + asel];
            acc[g] = wmma_bf16(af, bf, acc[g]);
        }
    }

    // Epilogue: D layout -> lanes 0-15 hold M=r, lanes 16-31 hold M=8+r; N = lane&15.
    const int mbase = m0 + wave * 16 + (lane >> 4) * 8;
    #pragma unroll
    for (int g = 0; g < 8; ++g) {
        int n = n0 + g * 16 + lrow;
        #pragma unroll
        for (int r = 0; r < 8; ++r) {
            float v = acc[g][r];
            if (STORE_F32) ((float*)C)[(size_t)(mbase + r) * N + n] = v;
            else           ((__bf16*)C)[(size_t)(mbase + r) * N + n] = (__bf16)v;
        }
    }
}

// ---------------------------------------------------------------------------
// RoPE in place on (SEQ, nheads, 128) bf16, interleaved pairs (2p, 2p+1).
// ---------------------------------------------------------------------------
__global__ void rope_inplace(__bf16* __restrict__ buf,
                             const float* __restrict__ cosT,
                             const float* __restrict__ sinT, int nheads) {
    int idx = blockIdx.x * blockDim.x + threadIdx.x;
    int total = SEQL * nheads * (HD / 2);
    if (idx >= total) return;
    int p = idx & 63;              // HD/2 = 64
    int rest = idx >> 6;
    int h = rest % nheads;
    int s = rest / nheads;
    float c = cosT[s * 64 + p], sn = sinT[s * 64 + p];
    __bf16* b = buf + ((size_t)s * nheads + h) * HD + 2 * p;
    float xr = (float)b[0], xi = (float)b[1];
    b[0] = (__bf16)(xr * c - xi * sn);
    b[1] = (__bf16)(xr * sn + xi * c);
}

// ---------------------------------------------------------------------------
// KV-cache fill: positions 0..SEQ-1 get roped K / raw V; tail gets zeros
// (cache starts as zeros in the reference; d_out is poisoned so we must write).
// ---------------------------------------------------------------------------
__global__ void fill_cache(const __bf16* __restrict__ kb,
                           const __bf16* __restrict__ vb,
                           float* __restrict__ ck, float* __restrict__ cv) {
    int idx = blockIdx.x * blockDim.x + threadIdx.x;
    const int total = WIN * KVDIM;
    if (idx >= total) return;
    int pos = idx >> 10;           // KVDIM = 1024
    float k = 0.f, v = 0.f;
    if (pos < SEQL) { k = (float)kb[idx]; v = (float)vb[idx]; }
    ck[idx] = k;
    cv[idx] = v;
}

// ---------------------------------------------------------------------------
// Flash attention: grid (SEQ/128, NH). Block = 8 waves; wave owns 16 q-rows.
// Key blocks of 32; online softmax; all matmuls via bf16 WMMA.
// ---------------------------------------------------------------------------
__global__ __launch_bounds__(256) void flash_attn(
    const __bf16* __restrict__ Q, const __bf16* __restrict__ K,
    const __bf16* __restrict__ V, __bf16* __restrict__ O)
{
    __shared__ __attribute__((aligned(16))) __bf16 sK[32][HD];       // 8 KB
    __shared__ __attribute__((aligned(16))) __bf16 sVt[HD][32];      // 8 KB (transposed)
    __shared__ __attribute__((aligned(16))) __bf16 sP[8][16][32];    // 8 KB, per-wave P tile

    const int h    = blockIdx.y;
    const int kvh  = h >> 2;                 // REPEATS = NH/NKV = 4
    const int q0b  = blockIdx.x * 128;
    const int tid  = threadIdx.x;
    const int wave = tid >> 5;
    const int lane = tid & 31;
    const int lrow = lane & 15;
    const int asel = (lane >> 4) * 8;
    const int bsel = (lane >> 4) * 16;
    const int qrow = q0b + wave * 16;

    // Q fragments (A layout), 4 chunks of K=32 along HD=128.
    BFrag qf[4];
    const __bf16* qp = Q + (size_t)(qrow + lrow) * QDIM + h * HD;
    #pragma unroll
    for (int c = 0; c < 4; ++c) {
        qf[c].u[0] = *(const uint4*)(qp + c * 32 + asel);
        qf[c].u[1] = *(const uint4*)(qp + c * 32 + 16 + asel);
    }

    v8f o[8] = {};
    float mi[8], li[8];
    #pragma unroll
    for (int r = 0; r < 8; ++r) { mi[r] = -1e30f; li[r] = 0.f; }

    const int nkb = (q0b >> 5) + 4;          // causal: keys up to q0b+128
    for (int kb = 0; kb < nkb; ++kb) {
        const int k0 = kb * 32;
        __syncthreads();
        {   // Cooperative stage of 32 keys x 128 dims of K (row-major) and V (transposed).
            int key = tid >> 3, ch = (tid & 7) * 16;
            const __bf16* kp = K + (size_t)(k0 + key) * KVDIM + kvh * HD + ch;
            const __bf16* vp = V + (size_t)(k0 + key) * KVDIM + kvh * HD + ch;
            *(uint4*)&sK[key][ch]     = *(const uint4*)(kp);
            *(uint4*)&sK[key][ch + 8] = *(const uint4*)(kp + 8);
            uint4 v0 = *(const uint4*)(vp), v1 = *(const uint4*)(vp + 8);
            const __bf16* ve = (const __bf16*)&v0;
            #pragma unroll
            for (int e = 0; e < 8; ++e) sVt[ch + e][key] = ve[e];
            ve = (const __bf16*)&v1;
            #pragma unroll
            for (int e = 0; e < 8; ++e) sVt[ch + 8 + e][key] = ve[e];
        }
        __syncthreads();

        // S = Q @ K^T : two 16-key column tiles.
        v8f s0 = {}, s1 = {};
        #pragma unroll
        for (int c = 0; c < 4; ++c) {
            BFrag kf0, kf1;                  // B layout: n = key (lane&15), K in table order (same as A)
            kf0.u[0] = *(const uint4*)&sK[lrow][c * 32 + asel];
            kf0.u[1] = *(const uint4*)&sK[lrow][c * 32 + 16 + asel];
            kf1.u[0] = *(const uint4*)&sK[16 + lrow][c * 32 + asel];
            kf1.u[1] = *(const uint4*)&sK[16 + lrow][c * 32 + 16 + asel];
            s0 = wmma_bf16(qf[c], kf0, s0);
            s1 = wmma_bf16(qf[c], kf1, s1);
        }

        // Mask + online softmax. Row m = (lane>>4)*8 + r lives in 16 lanes of one half.
        const int mrow0 = qrow + (lane >> 4) * 8;
        #pragma unroll
        for (int r = 0; r < 8; ++r) {
            int qi = mrow0 + r;
            float a0 = s0[r] * ATTN_SCALE, a1 = s1[r] * ATTN_SCALE;
            if (k0 + lrow      > qi) a0 = -1e30f;
            if (k0 + 16 + lrow > qi) a1 = -1e30f;
            float lm = fmaxf(a0, a1);
            lm = fmaxf(lm, __shfl_xor(lm, 1, 32));
            lm = fmaxf(lm, __shfl_xor(lm, 2, 32));
            lm = fmaxf(lm, __shfl_xor(lm, 4, 32));
            lm = fmaxf(lm, __shfl_xor(lm, 8, 32));
            float mnew  = fmaxf(mi[r], lm);
            float alpha = __expf(mi[r] - mnew);
            float p0 = __expf(a0 - mnew), p1 = __expf(a1 - mnew);
            float ls = p0 + p1;
            ls += __shfl_xor(ls, 1, 32);
            ls += __shfl_xor(ls, 2, 32);
            ls += __shfl_xor(ls, 4, 32);
            ls += __shfl_xor(ls, 8, 32);
            li[r] = li[r] * alpha + ls;
            mi[r] = mnew;
            #pragma unroll
            for (int g = 0; g < 8; ++g) o[g][r] = o[g][r] * alpha;
            int pr = (lane >> 4) * 8 + r;    // D-layout store of P (as bf16)
            sP[wave][pr][lrow]      = (__bf16)p0;
            sP[wave][pr][16 + lrow] = (__bf16)p1;
        }

        // O += P @ V. Per-wave LDS round-trip converts D layout -> A layout.
        // (DS ops are in-order within a wave; each wave touches only sP[wave].)
        BFrag pf;
        pf.u[0] = *(const uint4*)&sP[wave][lrow][asel];
        pf.u[1] = *(const uint4*)&sP[wave][lrow][16 + asel];
        #pragma unroll
        for (int g = 0; g < 8; ++g) {
            BFrag vf;                        // B layout: n = dim, K = keys in table order (same as A)
            vf.u[0] = *(const uint4*)&sVt[g * 16 + lrow][asel];
            vf.u[1] = *(const uint4*)&sVt[g * 16 + lrow][16 + asel];
            o[g] = wmma_bf16(pf, vf, o[g]);
        }
    }

    // Epilogue: normalize by row sum, store bf16 to (row, h*128 + dim).
    const int mrow0 = qrow + (lane >> 4) * 8;
    #pragma unroll
    for (int g = 0; g < 8; ++g) {
        #pragma unroll
        for (int r = 0; r < 8; ++r) {
            float val = o[g][r] / li[r];
            O[(size_t)(mrow0 + r) * QDIM + h * HD + g * 16 + lrow] = (__bf16)val;
        }
    }
}

// ---------------------------------------------------------------------------
// Launcher
// ---------------------------------------------------------------------------
extern "C" void kernel_launch(void* const* d_in, const int* in_sizes, int n_in,
                              void* d_out, int out_size, void* d_ws, size_t ws_size,
                              hipStream_t stream)
{
    const float* x  = (const float*)d_in[0];
    const float* wq = (const float*)d_in[1];
    const float* wk = (const float*)d_in[2];
    const float* wv = (const float*)d_in[3];
    const float* wo = (const float*)d_in[4];
    const float* fc = (const float*)d_in[5];
    const float* fs = (const float*)d_in[6];
    // d_in[7] positions (identity here), [8] mask (recomputed causally),
    // [9]/[10] cache_k/v inputs (zeros) are not needed.

    char* ws = (char*)d_ws;
    size_t off = 0;
    auto carve = [&](size_t bytes) { char* p = ws + off; off += bytes; return p; };
    __bf16* XB  = (__bf16*)carve((size_t)SEQL * DIMD * 2);   // x          16 MB
    __bf16* WQB = (__bf16*)carve((size_t)QDIM * DIMD * 2);   // wq         32 MB
    __bf16* WKB = (__bf16*)carve((size_t)KVDIM * DIMD * 2);  // wk          8 MB
    __bf16* WVB = (__bf16*)carve((size_t)KVDIM * DIMD * 2);  // wv          8 MB
    __bf16* WOB = (__bf16*)carve((size_t)DIMD * QDIM * 2);   // wo         32 MB
    __bf16* QB  = (__bf16*)carve((size_t)SEQL * QDIM * 2);   // q          16 MB
    __bf16* KB  = (__bf16*)carve((size_t)SEQL * KVDIM * 2);  // k           4 MB
    __bf16* VB  = (__bf16*)carve((size_t)SEQL * KVDIM * 2);  // v           4 MB
    __bf16* AB  = (__bf16*)carve((size_t)SEQL * QDIM * 2);   // attn out   16 MB
    (void)ws_size;

    auto cvt = [&](const float* src, __bf16* dst, size_t n) {
        int n4 = (int)(n / 4);
        cvt_f32_to_bf16<<<(n4 + 255) / 256, 256, 0, stream>>>(src, dst, n4);
    };
    cvt(x,  XB,  (size_t)SEQL * DIMD);
    cvt(wq, WQB, (size_t)QDIM * DIMD);
    cvt(wk, WKB, (size_t)KVDIM * DIMD);
    cvt(wv, WVB, (size_t)KVDIM * DIMD);
    cvt(wo, WOB, (size_t)DIMD * QDIM);

    // QKV projections (bf16 out)
    gemm_bf16_xwT<false><<<dim3(SEQL / 128, QDIM / 128), 256, 0, stream>>>(
        XB, WQB, QB, SEQL, QDIM, DIMD);
    gemm_bf16_xwT<false><<<dim3(SEQL / 128, KVDIM / 128), 256, 0, stream>>>(
        XB, WKB, KB, SEQL, KVDIM, DIMD);
    gemm_bf16_xwT<false><<<dim3(SEQL / 128, KVDIM / 128), 256, 0, stream>>>(
        XB, WVB, VB, SEQL, KVDIM, DIMD);

    // RoPE (q and k)
    {
        int tq = SEQL * NH * (HD / 2);
        rope_inplace<<<(tq + 255) / 256, 256, 0, stream>>>(QB, fc, fs, NH);
        int tk = SEQL * NKV * (HD / 2);
        rope_inplace<<<(tk + 255) / 256, 256, 0, stream>>>(KB, fc, fs, NKV);
    }

    // Output tuple layout: [out (2048*4096) | cache_k (4096*1024) | cache_v (4096*1024)]
    float* outF = (float*)d_out;
    float* ck = outF + (size_t)SEQL * DIMD;
    float* cv = ck + (size_t)WIN * KVDIM;
    {
        int tot = WIN * KVDIM;
        fill_cache<<<(tot + 255) / 256, 256, 0, stream>>>(KB, VB, ck, cv);
    }

    // Attention
    flash_attn<<<dim3(SEQL / 128, NH), 256, 0, stream>>>(QB, KB, VB, AB);

    // Output projection (f32 out, straight into d_out)
    gemm_bf16_xwT<true><<<dim3(SEQL / 128, DIMD / 128), 256, 0, stream>>>(
        AB, WOB, outF, SEQL, DIMD, QDIM);
}
